// Sequence_67465346285932
// MI455X (gfx1250) — matrix operand order repaired and checked by reading, not compiled, not run on hardware
//
#include <hip/hip_runtime.h>
#include <stdint.h>

// Problem dims (fixed by reference)
#define B_ 128
#define T_ 512
#define V_ 512
#define H_ 1024
#define E_ 512
#define C_ 128
#define R_ 512
#define G4 (4 * H_)     // 4096 gate width
#define KC (E_ + C_)    // 640 concat(embed, cond) width
#define KCH 128         // K-chunk staged into LDS in the recurrent kernel

typedef __attribute__((ext_vector_type(16))) __bf16          bf16x16;
typedef __attribute__((ext_vector_type(8)))  float           f32x8;
typedef __attribute__((ext_vector_type(8)))  unsigned short  u16x8;

union FragU { bf16x16 v; u16x8 h[2]; };

__device__ __forceinline__ unsigned short f2bf(float f) {
  unsigned x = __builtin_bit_cast(unsigned, f);
  unsigned r = x + 0x7FFFu + ((x >> 16) & 1u);   // round-to-nearest-even
  return (unsigned short)(r >> 16);
}
__device__ __forceinline__ float sigm(float x) { return 1.0f / (1.0f + __expf(-x)); }

// ---------------- prep / elementwise kernels ----------------

__global__ void zero_u32_kernel(unsigned* __restrict__ p, long n) {
  long i = (long)blockIdx.x * blockDim.x + threadIdx.x;
  if (i < n) p[i] = 0u;
}

__global__ void cvt_f32_bf16_kernel(const float* __restrict__ s,
                                    unsigned short* __restrict__ d, long n) {
  long i = (long)blockIdx.x * blockDim.x + threadIdx.x;
  if (i < n) d[i] = f2bf(s[i]);
}

__global__ void vec_add_kernel(const float* __restrict__ a, const float* __restrict__ b,
                               float* __restrict__ o, int n) {
  int i = blockIdx.x * blockDim.x + threadIdx.x;
  if (i < n) o[i] = a[i] + b[i];
}

// Wcond[R, 640] = concat_k(ee_W[R,512], ce_W[R,128]) as bf16; cbias = ee_b + ce_b
__global__ void build_wcond_kernel(const float* __restrict__ eeW, const float* __restrict__ ceW,
                                   const float* __restrict__ eeb, const float* __restrict__ ceb,
                                   unsigned short* __restrict__ wc, float* __restrict__ cb) {
  long i = (long)blockIdx.x * blockDim.x + threadIdx.x;
  if (i < (long)R_ * KC) {
    int r = (int)(i / KC), col = (int)(i % KC);
    float v = (col < E_) ? eeW[(long)r * E_ + col] : ceW[(long)r * C_ + (col - E_)];
    wc[i] = f2bf(v);
  }
  if (i < R_) cb[i] = eeb[i] + ceb[i];
}

// Ain[B*T, 640] = concat(embed_W[input_ids[b,t]], features[b,t]) as bf16
__global__ void gather_concat_kernel(const int* __restrict__ ids, const float* __restrict__ feat,
                                     const float* __restrict__ embW,
                                     unsigned short* __restrict__ Ain) {
  long i = (long)blockIdx.x * blockDim.x + threadIdx.x;
  const long n = (long)B_ * T_ * KC;
  if (i >= n) return;
  long row = i / KC; int col = (int)(i % KC);
  float v = (col < E_) ? embW[(long)ids[row] * E_ + col]
                       : feat[row * (long)C_ + (col - E_)];
  Ain[i] = f2bf(v);
}

// ---------------- WMMA GEMM: C[M,N] = A[M,K] * W[N,K]^T + bias[N] ----------------
// Block: 256 thr = 8 waves. Block tile 128x16, each wave one 16x16 WMMA tile.
// A,W bf16 row-major; C fp32 or bf16.  (Compiler pipelines the global loads well;
// kept on the direct-global path.)
__global__ __launch_bounds__(256) void gemm_bf16_wt_kernel(
    const unsigned short* __restrict__ A, long lda,
    const unsigned short* __restrict__ W, long ldw,
    const float* __restrict__ bias,
    void* __restrict__ Cout, long ldc, int out_bf16, int K)
{
  const int lane = threadIdx.x & 31;
  const int wid  = threadIdx.x >> 5;
  const long m0 = (long)blockIdx.y * 128 + (long)wid * 16;
  const long n0 = (long)blockIdx.x * 16;
  const int sub  = lane & 15;
  const int offA = (lane < 16) ? 0 : 8;    // A: lanes 0-15 K{0..7,16..23}, 16-31 K{8..15,24..31}
  const int offB = (lane < 16) ? 0 : 8;    // B: lanes 0-15 K{0..7,16..23}, 16-31 K{8..15,24..31}
  const unsigned short* Arow = A + (m0 + sub) * lda;
  const unsigned short* Wrow = W + (n0 + sub) * ldw;

  f32x8 acc = {0.f, 0.f, 0.f, 0.f, 0.f, 0.f, 0.f, 0.f};
  for (int k = 0; k < K; k += 64) {   // all K here are multiples of 64 (640,512,1024)
    FragU a0, b0, a1, b1;
    a0.h[0] = *reinterpret_cast<const u16x8*>(Arow + k + offA);
    a0.h[1] = *reinterpret_cast<const u16x8*>(Arow + k + 16 + offA);
    b0.h[0] = *reinterpret_cast<const u16x8*>(Wrow + k + offB);
    b0.h[1] = *reinterpret_cast<const u16x8*>(Wrow + k + offB + 16);
    a1.h[0] = *reinterpret_cast<const u16x8*>(Arow + k + 32 + offA);
    a1.h[1] = *reinterpret_cast<const u16x8*>(Arow + k + 48 + offA);
    b1.h[0] = *reinterpret_cast<const u16x8*>(Wrow + k + 32 + offB);
    b1.h[1] = *reinterpret_cast<const u16x8*>(Wrow + k + 32 + offB + 16);
    __builtin_prefetch(Arow + k + 128, 0, 1);
    __builtin_prefetch(Wrow + k + 128, 0, 1);
    acc = __builtin_amdgcn_wmma_f32_16x16x32_bf16(false, a0.v, false, b0.v,
                                                  (short)0, acc, false, false);
    acc = __builtin_amdgcn_wmma_f32_16x16x32_bf16(false, a1.v, false, b1.v,
                                                  (short)0, acc, false, false);
  }

  const long n = n0 + sub;
  const float bv = bias ? bias[n] : 0.0f;
  const int mhi = (lane >> 4) * 8;   // C/D: VGPR r -> M = r + 8*(lane>=16)
#pragma unroll
  for (int r = 0; r < 8; ++r) {
    long m = m0 + mhi + r;
    float v = acc[r] + bv;
    if (out_bf16) ((unsigned short*)Cout)[m * ldc + n] = f2bf(v);
    else          ((float*)Cout)[m * ldc + n] = v;
  }
}

// ---------------- fused LSTM recurrent step ----------------
// gates[b, g*H+n] = Gin_t[b, g*H+n] + sum_k hprev[b,k] * Whh[g*H+n, k]
// W_hh tile (4 gates x 16 cols) is shared by all 8 waves -> staged through LDS
// with CDNA5 async global->LDS copies (ASYNCcnt) for 8x reuse; batch activations
// (A fragments, private per wave) stay on pipelined global loads.
// grid = H/16 = 64 blocks, 256 threads.
__global__ __launch_bounds__(256) void lstm_step_kernel(
    const unsigned short* __restrict__ hprev,   // [B,H] bf16
    const unsigned short* __restrict__ Whh,     // [4H,H] bf16
    const float* __restrict__ gin_t,            // Gin + t*4H; batch stride T*4H
    float* __restrict__ cstate,                 // [B,H] fp32 in/out
    unsigned short* __restrict__ hnext,         // [B,H] bf16 (ping-pong)
    unsigned short* __restrict__ hseq_t)        // Hseq + t*H; batch stride T*H
{
  __shared__ unsigned short ldsW[64 * KCH];     // 16 KB: rows = gate*16+col, k in chunk

  const int tid  = threadIdx.x;
  const int lane = tid & 31;
  const int wid  = tid >> 5;
  const int m0 = wid * 16;                  // 8 waves * 16 = B = 128
  const int n0 = blockIdx.x * 16;
  const int sub  = lane & 15;
  const int offA = (lane < 16) ? 0 : 8;
  const int offB = (lane < 16) ? 0 : 8;

  const unsigned short* Arow = hprev + (long)(m0 + sub) * H_;

  f32x8 ai = {0.f,0.f,0.f,0.f,0.f,0.f,0.f,0.f};
  f32x8 af = ai, ag = ai, ao = ai;

  for (int kc = 0; kc < H_; kc += KCH) {
    // Stage 64 rows x KCH of Whh into LDS: 1024 x 16B transfers, 4 per thread,
    // issued as async copies (no VGPR data return, tracked with ASYNCcnt).
#pragma unroll
    for (int u = 0; u < 4; ++u) {
      int unit = tid + u * 256;             // 0..1023
      int row  = unit >> 4;                 // 0..63  (gate = row>>4, col = row&15)
      int kin  = (unit & 15) * 8;           // 0..120 (ushort units, 16B aligned)
      const unsigned short* gp =
          Whh + (long)((row >> 4) * H_ + n0 + (row & 15)) * H_ + kc + kin;
      unsigned lds_off = (unsigned)(uintptr_t)(&ldsW[row * KCH + kin]);
      unsigned long long ga = (unsigned long long)(uintptr_t)gp;
      asm volatile("global_load_async_to_lds_b128 %0, %1, off"
                   :: "v"(lds_off), "v"(ga) : "memory");
    }
    asm volatile("s_wait_asynccnt 0" ::: "memory");
    __syncthreads();

#pragma unroll
    for (int kk = 0; kk < KCH; kk += 32) {
      FragU a, bi, bf, bg, bo;
      const int k = kc + kk;
      a.h[0]  = *reinterpret_cast<const u16x8*>(Arow + k + offA);
      a.h[1]  = *reinterpret_cast<const u16x8*>(Arow + k + 16 + offA);
      const unsigned short* L0 = &ldsW[(0 * 16 + sub) * KCH + kk + offB];
      const unsigned short* L1 = &ldsW[(1 * 16 + sub) * KCH + kk + offB];
      const unsigned short* L2 = &ldsW[(2 * 16 + sub) * KCH + kk + offB];
      const unsigned short* L3 = &ldsW[(3 * 16 + sub) * KCH + kk + offB];
      bi.h[0] = *reinterpret_cast<const u16x8*>(L0);
      bi.h[1] = *reinterpret_cast<const u16x8*>(L0 + 16);
      bf.h[0] = *reinterpret_cast<const u16x8*>(L1);
      bf.h[1] = *reinterpret_cast<const u16x8*>(L1 + 16);
      bg.h[0] = *reinterpret_cast<const u16x8*>(L2);
      bg.h[1] = *reinterpret_cast<const u16x8*>(L2 + 16);
      bo.h[0] = *reinterpret_cast<const u16x8*>(L3);
      bo.h[1] = *reinterpret_cast<const u16x8*>(L3 + 16);
      ai = __builtin_amdgcn_wmma_f32_16x16x32_bf16(false, a.v, false, bi.v, (short)0, ai, false, false);
      af = __builtin_amdgcn_wmma_f32_16x16x32_bf16(false, a.v, false, bf.v, (short)0, af, false, false);
      ag = __builtin_amdgcn_wmma_f32_16x16x32_bf16(false, a.v, false, bg.v, (short)0, ag, false, false);
      ao = __builtin_amdgcn_wmma_f32_16x16x32_bf16(false, a.v, false, bo.v, (short)0, ao, false, false);
    }
    __syncthreads();   // protect LDS before next chunk overwrites it
  }

  const int n = n0 + sub;
  const int mhi = (lane >> 4) * 8;
#pragma unroll
  for (int r = 0; r < 8; ++r) {
    int m = m0 + mhi + r;
    const float* g = gin_t + (long)m * ((long)T_ * G4);
    float iv = sigm(ai[r] + g[0 * H_ + n]);
    float fv = sigm(af[r] + g[1 * H_ + n]);
    float gv = tanhf(ag[r] + g[2 * H_ + n]);
    float ov = sigm(ao[r] + g[3 * H_ + n]);
    long ci = (long)m * H_ + n;
    float cv = fv * cstate[ci] + iv * gv;
    cstate[ci] = cv;
    float hv = ov * tanhf(cv);
    unsigned short hb = f2bf(hv);
    hnext[ci] = hb;
    hseq_t[(long)m * ((long)T_ * H_) + n] = hb;
  }
}

// ---------------- host orchestration ----------------

extern "C" void kernel_launch(void* const* d_in, const int* in_sizes, int n_in,
                              void* d_out, int out_size, void* d_ws, size_t ws_size,
                              hipStream_t stream) {
  const int*   ids  = (const int*)d_in[0];
  const float* feat = (const float*)d_in[1];
  const float* embW = (const float*)d_in[2];
  const float* eeW  = (const float*)d_in[3];
  const float* eeb  = (const float*)d_in[4];
  const float* ceW  = (const float*)d_in[5];
  const float* ceb  = (const float*)d_in[6];
  const float* Wih1 = (const float*)d_in[7];
  const float* Whh1 = (const float*)d_in[8];
  const float* bih1 = (const float*)d_in[9];
  const float* bhh1 = (const float*)d_in[10];
  const float* Wih2 = (const float*)d_in[11];
  const float* Whh2 = (const float*)d_in[12];
  const float* bih2 = (const float*)d_in[13];
  const float* bhh2 = (const float*)d_in[14];
  const float* linW = (const float*)d_in[15];
  const float* linb = (const float*)d_in[16];
  float* out = (float*)d_out;

  char* ws = (char*)d_ws;
  size_t off = 0;
  auto alloc = [&](size_t bytes) -> char* {
    char* p = ws + off;
    off = (off + bytes + 255) & ~(size_t)255;
    return p;
  };
  unsigned short* wcond = (unsigned short*)alloc((size_t)R_ * KC * 2);
  float*          cbias = (float*)alloc((size_t)R_ * 4);
  unsigned short* wih1b = (unsigned short*)alloc((size_t)G4 * R_ * 2);
  unsigned short* whh1b = (unsigned short*)alloc((size_t)G4 * H_ * 2);
  unsigned short* wih2b = (unsigned short*)alloc((size_t)G4 * H_ * 2);
  unsigned short* whh2b = (unsigned short*)alloc((size_t)G4 * H_ * 2);
  unsigned short* linwb = (unsigned short*)alloc((size_t)V_ * H_ * 2);
  float*          b1    = (float*)alloc((size_t)G4 * 4);
  float*          b2    = (float*)alloc((size_t)G4 * 4);
  unsigned short* Ain   = (unsigned short*)alloc((size_t)B_ * T_ * KC * 2);
  unsigned short* X     = (unsigned short*)alloc((size_t)B_ * T_ * R_ * 2);
  float*          Gin   = (float*)alloc((size_t)B_ * T_ * G4 * 4);   // reused for both layers
  unsigned short* H1    = (unsigned short*)alloc((size_t)B_ * T_ * H_ * 2);
  unsigned short* H2    = (unsigned short*)alloc((size_t)B_ * T_ * H_ * 2);
  unsigned short* h1p   = (unsigned short*)alloc((size_t)2 * B_ * H_ * 2);
  unsigned short* h2p   = (unsigned short*)alloc((size_t)2 * B_ * H_ * 2);
  float*          c1    = (float*)alloc((size_t)B_ * H_ * 4);
  float*          c2    = (float*)alloc((size_t)B_ * H_ * 4);

  auto grid1 = [](long n) { return dim3((unsigned)((n + 255) / 256)); };

  // Weight conversion to bf16 (one-time; all weights then resident in L2).
  cvt_f32_bf16_kernel<<<grid1((long)G4 * R_), 256, 0, stream>>>(Wih1, wih1b, (long)G4 * R_);
  cvt_f32_bf16_kernel<<<grid1((long)G4 * H_), 256, 0, stream>>>(Whh1, whh1b, (long)G4 * H_);
  cvt_f32_bf16_kernel<<<grid1((long)G4 * H_), 256, 0, stream>>>(Wih2, wih2b, (long)G4 * H_);
  cvt_f32_bf16_kernel<<<grid1((long)G4 * H_), 256, 0, stream>>>(Whh2, whh2b, (long)G4 * H_);
  cvt_f32_bf16_kernel<<<grid1((long)V_ * H_), 256, 0, stream>>>(linW, linwb, (long)V_ * H_);
  build_wcond_kernel<<<grid1((long)R_ * KC), 256, 0, stream>>>(eeW, ceW, eeb, ceb, wcond, cbias);
  vec_add_kernel<<<grid1(G4), 256, 0, stream>>>(bih1, bhh1, b1, G4);
  vec_add_kernel<<<grid1(G4), 256, 0, stream>>>(bih2, bhh2, b2, G4);

  // Zero initial h (ping slot 0) and c states (ws is poisoned by harness).
  zero_u32_kernel<<<grid1((long)B_ * H_ / 2), 256, 0, stream>>>((unsigned*)h1p, (long)B_ * H_ / 2);
  zero_u32_kernel<<<grid1((long)B_ * H_ / 2), 256, 0, stream>>>((unsigned*)h2p, (long)B_ * H_ / 2);
  zero_u32_kernel<<<grid1((long)B_ * H_), 256, 0, stream>>>((unsigned*)c1, (long)B_ * H_);
  zero_u32_kernel<<<grid1((long)B_ * H_), 256, 0, stream>>>((unsigned*)c2, (long)B_ * H_);

  // Ain = concat(embed gather, features) in bf16
  gather_concat_kernel<<<grid1((long)B_ * T_ * KC), 256, 0, stream>>>(ids, feat, embW, Ain);

  // X[B*T,R] = Ain @ Wcond^T + (ee_b+ce_b)   (bf16 out; feeds next GEMM)
  gemm_bf16_wt_kernel<<<dim3(R_ / 16, (B_ * T_) / 128), 256, 0, stream>>>(
      Ain, KC, wcond, KC, cbias, X, R_, 1, KC);

  // Gin[B*T,4H] = X @ W_ih1^T + (b_ih1+b_hh1)   (hoisted layer-1 input projection)
  gemm_bf16_wt_kernel<<<dim3(G4 / 16, (B_ * T_) / 128), 256, 0, stream>>>(
      X, R_, wih1b, R_, b1, Gin, G4, 0, R_);

  // Layer-1 recurrence (sequential over T; W_hh1 stays hot in L2)
  for (int t = 0; t < T_; ++t) {
    lstm_step_kernel<<<dim3(H_ / 16), 256, 0, stream>>>(
        h1p + (size_t)(t & 1) * B_ * H_, whh1b,
        Gin + (size_t)t * G4, c1,
        h1p + (size_t)((t + 1) & 1) * B_ * H_,
        H1 + (size_t)t * H_);
  }

  // Gin[B*T,4H] = H1 @ W_ih2^T + (b_ih2+b_hh2)   (hoisted layer-2 input projection)
  gemm_bf16_wt_kernel<<<dim3(G4 / 16, (B_ * T_) / 128), 256, 0, stream>>>(
      H1, H_, wih2b, H_, b2, Gin, G4, 0, H_);

  // Layer-2 recurrence
  for (int t = 0; t < T_; ++t) {
    lstm_step_kernel<<<dim3(H_ / 16), 256, 0, stream>>>(
        h2p + (size_t)(t & 1) * B_ * H_, whh2b,
        Gin + (size_t)t * G4, c2,
        h2p + (size_t)((t + 1) & 1) * B_ * H_,
        H2 + (size_t)t * H_);
  }

  // out[B,T,V] = H2 @ lin_W^T + lin_b
  gemm_bf16_wt_kernel<<<dim3(V_ / 16, (B_ * T_) / 128), 256, 0, stream>>>(
      H2, H_, linwb, H_, linb, out, V_, 0, H_);

  (void)in_sizes; (void)n_in; (void)out_size; (void)ws_size;
}
